// LogicGatedSNN_1846835937527
// MI455X (gfx1250) — 8-bit matrix operand order repaired and checked by reading, compiled, not run on hardware
//
#include <hip/hip_runtime.h>

typedef __attribute__((ext_vector_type(16))) int   v16i;
typedef __attribute__((ext_vector_type(8)))  float v8f;
typedef __attribute__((ext_vector_type(4)))  int   v4i;
typedef __attribute__((ext_vector_type(2)))  int   v2i;

#define FP8_POS_ONE 0x38u
#define FP8_NEG_ONE 0xB8u

__device__ __forceinline__ unsigned tern_fp8(float x) {
  return x > 1.0f ? FP8_POS_ONE : (x < -1.0f ? FP8_NEG_ONE : 0u);
}
__device__ __forceinline__ unsigned spike_fp8(float x) {
  return (x != 0.0f) ? FP8_POS_ONE : 0u;
}

__device__ __forceinline__ unsigned pack4_tern(float4 f) {
  return tern_fp8(f.x) | (tern_fp8(f.y) << 8) | (tern_fp8(f.z) << 16) | (tern_fp8(f.w) << 24);
}
__device__ __forceinline__ unsigned pack4_spike(float4 f) {
  return spike_fp8(f.x) | (spike_fp8(f.y) << 8) | (spike_fp8(f.z) << 16) | (spike_fp8(f.w) << 24);
}

__global__ __launch_bounds__(256) void quant_w_kernel(const float* __restrict__ s,
                                                      uint4* __restrict__ q, int n16) {
  int i = blockIdx.x * blockDim.x + threadIdx.x;
  if (i >= n16) return;
  const float4* sp = (const float4*)s + (size_t)i * 4;
  float4 f0 = sp[0], f1 = sp[1], f2 = sp[2], f3 = sp[3];
  uint4 o;
  o.x = pack4_tern(f0); o.y = pack4_tern(f1);
  o.z = pack4_tern(f2); o.w = pack4_tern(f3);
  q[i] = o;
}

__global__ __launch_bounds__(256) void quant_s_kernel(const float* __restrict__ s,
                                                      uint4* __restrict__ q, int n16) {
  int i = blockIdx.x * blockDim.x + threadIdx.x;
  if (i >= n16) return;
  const float4* sp = (const float4*)s + (size_t)i * 4;
  float4 f0 = sp[0], f1 = sp[1], f2 = sp[2], f3 = sp[3];
  uint4 o;
  o.x = pack4_spike(f0); o.y = pack4_spike(f1);
  o.z = pack4_spike(f2); o.w = pack4_spike(f3);
  q[i] = o;
}

__global__ __launch_bounds__(256) void snn_wmma_kernel(
    const unsigned char* __restrict__ Aq,
    const unsigned char* __restrict__ Wq,
    const float* __restrict__ vmem,
    const float* __restrict__ thr,
    const float* __restrict__ refr,
    float* __restrict__ out,
    int Bn, int IN, int OUT) {
  const int lane = threadIdx.x & 31;
  const int wave = threadIdx.x >> 5;
  const int wm = wave & 1;
  const int wn = wave >> 1;
  const int h  = lane >> 4;
  const int l  = lane & 15;

  const int m_base = blockIdx.x * 64  + wm * 32;
  const int n_base = blockIdx.y * 256 + wn * 64;

  v8f acc[2][4] = {};

  const unsigned char* arow0 = Aq + (size_t)(m_base + l)      * IN;
  const unsigned char* arow1 = Aq + (size_t)(m_base + 16 + l) * IN;
  const unsigned char* brow0 = Wq + (size_t)(n_base + l)      * IN;

  for (int k0 = 0; k0 < IN; k0 += 128) {
    if (k0 + 128 < IN) {
      __builtin_prefetch(arow0 + k0 + 128, 0, 3);
      __builtin_prefetch(arow1 + k0 + 128, 0, 3);
      __builtin_prefetch(brow0 + k0 + 128, 0, 3);
    }

    v16i a[2];
    {
      v2i* av0 = (v2i*)&a[0];
      v2i* av1 = (v2i*)&a[1];
#pragma unroll
      for (int c = 0; c < 8; ++c) {
        int koff = k0 + 32 * (c >> 1) + 16 * h + 8 * (c & 1);
        av0[c] = *(const v2i*)(arow0 + koff);
        av1[c] = *(const v2i*)(arow1 + koff);
      }
    }

    v16i b[4];
#pragma unroll
    for (int nt = 0; nt < 4; ++nt) {
      const unsigned char* brow = Wq + (size_t)(n_base + nt * 16 + l) * IN + k0;
      v4i* bv = (v4i*)&b[nt];
#pragma unroll
      for (int g = 0; g < 4; ++g) {
        bv[g] = *(const v4i*)(brow + 32 * g + 16 * h);
      }
    }

#pragma unroll
    for (int mt = 0; mt < 2; ++mt)
#pragma unroll
      for (int nt = 0; nt < 4; ++nt)
        acc[mt][nt] = __builtin_amdgcn_wmma_f32_16x16x128_fp8_fp8(
            a[mt], b[nt], (short)0, acc[mt][nt],
             false,  false);
  }

#pragma unroll
  for (int nt = 0; nt < 4; ++nt) {
    const int o = n_base + nt * 16 + l;
    const float gate = (refr[o] > 0.0f) ? 0.0f : 1.0f;
    const float leak = vmem[o] * 0.7f;
    const float t    = thr[o];
#pragma unroll
    for (int mt = 0; mt < 2; ++mt) {
      const int mrow = m_base + mt * 16 + 8 * h;
#pragma unroll
      for (int r = 0; r < 8; ++r) {
        float v = leak + acc[mt][nt][r] * gate;
        out[(size_t)(mrow + r) * OUT + o] = (v >= t) ? 1.0f : 0.0f;
      }
    }
  }
}

extern "C" void kernel_launch(void* const* d_in, const int* in_sizes, int n_in,
                              void* d_out, int out_size, void* d_ws, size_t ws_size,
                              hipStream_t stream) {
  const float* spike = (const float*)d_in[0];
  const float* syn   = (const float*)d_in[1];
  const float* vmem  = (const float*)d_in[2];
  const float* thr   = (const float*)d_in[3];
  const float* refr  = (const float*)d_in[4];

  const long long nS = in_sizes[0];
  const long long nW = in_sizes[1];
  const int OUT = in_sizes[2];
  const int IN  = (int)(nW / OUT);
  const int Bn  = (int)(nS / IN);

  unsigned char* sq = (unsigned char*)d_ws;
  unsigned char* wq = sq + nS;

  const int nS16 = (int)(nS / 16);
  const int nW16 = (int)(nW / 16);
  quant_s_kernel<<<(nS16 + 255) / 256, 256, 0, stream>>>(spike, (uint4*)sq, nS16);
  quant_w_kernel<<<(nW16 + 255) / 256, 256, 0, stream>>>(syn, (uint4*)wq, nW16);

  dim3 grid(Bn / 64, OUT / 256);
  snn_wmma_kernel<<<grid, 256, 0, stream>>>(sq, wq, vmem, thr, refr,
                                            (float*)d_out, Bn, IN, OUT);
}
